// CausalMultiHeadAttention_3478923510049
// MI455X (gfx1250) — matrix operand order repaired and checked by reading, not compiled, not run on hardware
//
#include <hip/hip_runtime.h>

typedef unsigned short u16;
typedef unsigned int   u32;
typedef unsigned long long u64;
typedef __attribute__((ext_vector_type(16))) __bf16 v16bf;
typedef __attribute__((ext_vector_type(8)))  float  v8f;
typedef __attribute__((ext_vector_type(4)))  u32    v4u;
typedef __attribute__((ext_vector_type(8)))  int    v8i;
typedef __attribute__((ext_vector_type(4)))  int    v4i;

union Frag16 { v16bf bf; v4u q[2]; };

constexpr int kB  = 2;
constexpr int kS  = 2048;
constexpr int kD  = 1024;
constexpr int kH  = 16;
constexpr int kDK = 64;
constexpr int kE  = 3 * kD;   // 3072
constexpr int kBS = kB * kS;  // 4096

#if defined(__has_builtin)
#  if __has_builtin(__builtin_amdgcn_tensor_load_to_lds)
#    define HAVE_TDM 1
#  endif
#endif
#ifndef HAVE_TDM
#  define HAVE_TDM 0
#endif

#if defined(__has_builtin)
#  if __has_builtin(__builtin_amdgcn_s_wait_tensorcnt)
#    define WAIT_TENSOR(n) __builtin_amdgcn_s_wait_tensorcnt(n)
#  endif
#endif
#ifndef WAIT_TENSOR
#  define WAIT_TENSOR(n) do {} while (0)
#endif

__device__ __forceinline__ u16 f2bf(float f) {
  u32 u = __float_as_uint(f);
  u += 0x7FFFu + ((u >> 16) & 1u);   // round-to-nearest-even
  return (u16)(u >> 16);
}

__device__ __forceinline__ v4u ld4(const u16* p) { return *(const v4u*)p; }

#define WMMA_BF16(A, B, C) \
  __builtin_amdgcn_wmma_f32_16x16x32_bf16(false, (A).bf, false, (B).bf, (short)0, (C), false, false)

// ---- TDM: 2D tile (bf16) global -> LDS, D# built per CDNA5 ISA §8 --------
#if HAVE_TDM
__device__ __forceinline__ void tdm_load_tile(u32 lds_byte_off, const u16* g,
                                              u32 td0, u32 td1, u32 st0,
                                              u32 tile0, u32 tile1) {
  u64 ga = (u64)(uintptr_t)g;
  v4u g0;
  g0[0] = 1u;                                                   // count=1, user desc
  g0[1] = lds_byte_off;                                         // lds_addr (bytes)
  g0[2] = (u32)ga;                                              // global_addr[95:64]
  g0[3] = (u32)((ga >> 32) & 0x01FFFFFFu) | 0x80000000u;        // addr[120:96] | type=2
  v8i g1;
  g1[0] = 0x00010000;                                           // data_size=1 (2 bytes)
  g1[1] = (int)((td0 & 0xFFFFu) << 16);                         // tensor_dim0[15:0]
  g1[2] = (int)(((td0 >> 16) & 0xFFFFu) | ((td1 & 0xFFFFu) << 16));
  g1[3] = (int)(((td1 >> 16) & 0xFFFFu) | ((tile0 & 0xFFFFu) << 16));
  g1[4] = (int)(tile1 & 0xFFFFu);                               // tile_dim1, tile_dim2=0
  g1[5] = (int)st0;                                             // tensor_dim0_stride[31:0]
  g1[6] = 0;
  g1[7] = 0;
  v4i z = {0, 0, 0, 0};
#if __clang_major__ >= 23
  v8i z8 = {0, 0, 0, 0, 0, 0, 0, 0};
  __builtin_amdgcn_tensor_load_to_lds(g0, g1, z, z, z8, 0);
#else
  __builtin_amdgcn_tensor_load_to_lds(g0, g1, z, z, 0);
#endif
}
#endif

// ---------------------------------------------------------------- convert
__global__ void convert_kernel(const float* __restrict__ X,
                               const float* __restrict__ Wqkv,
                               const float* __restrict__ Wout,
                               u16* __restrict__ Xb, u16* __restrict__ Wqb,
                               u16* __restrict__ Wob) {
  int i = blockIdx.x * blockDim.x + threadIdx.x;
  const int N1 = kBS * kD;
  const int N2 = 3 * kD * kD;
  const int N3 = kD * kD;
  if (i < N1)                 Xb[i]            = f2bf(X[i]);
  else if (i < N1 + N2)       Wqb[i - N1]      = f2bf(Wqkv[i - N1]);
  else if (i < N1 + N2 + N3)  Wob[i - N1 - N2] = f2bf(Wout[i - N1 - N2]);
}

// ---------------------------------------------------------------- GEMM core
// 32x64-per-wave bf16 GEMM, double-buffered over K.
// A: [M x kD] row-major bf16, W: [N x kD] row-major bf16 (B^T form).
struct GemmFrags {
  Frag16 a[2];
  Frag16 b[4];
};

__device__ __forceinline__ void gemm_load(GemmFrags& f, const u16* arow,
                                          const u16* Wb, int n0, int l, int lh,
                                          int k0) {
#pragma unroll
  for (int mt = 0; mt < 2; ++mt) {
    const u16* ap = arow + mt * 16 * kD + k0 + lh * 8;
    f.a[mt].q[0] = ld4(ap);
    f.a[mt].q[1] = ld4(ap + 16);
  }
#pragma unroll
  for (int j = 0; j < 4; ++j) {
    const u16* bp = Wb + (n0 + j * 16 + l) * kD + k0 + lh * 8;
    f.b[j].q[0] = ld4(bp);
    f.b[j].q[1] = ld4(bp + 16);
  }
}

__device__ __forceinline__ void gemm_mac(v8f acc[2][4], const GemmFrags& f) {
#pragma unroll
  for (int mt = 0; mt < 2; ++mt)
#pragma unroll
    for (int j = 0; j < 4; ++j)
      acc[mt][j] = WMMA_BF16(f.a[mt], f.b[j], acc[mt][j]);
}

__device__ __forceinline__ void gemm_body(v8f acc[2][4], const u16* arow,
                                          const u16* Wb, int n0, int l, int lh) {
  GemmFrags f0, f1;
  gemm_load(f0, arow, Wb, n0, l, lh, 0);
  for (int k0 = 0; k0 < kD - 64; k0 += 64) {
    gemm_load(f1, arow, Wb, n0, l, lh, k0 + 32);
    gemm_mac(acc, f0);
    gemm_load(f0, arow, Wb, n0, l, lh, k0 + 64);
    gemm_mac(acc, f1);
  }
  gemm_load(f1, arow, Wb, n0, l, lh, kD - 32);
  gemm_mac(acc, f0);
  gemm_mac(acc, f1);
}

// ------------------------------------------------------------- qkv GEMM
__global__ __launch_bounds__(128) void gemm_qkv_kernel(
    const u16* __restrict__ Xb, const u16* __restrict__ Wb,
    u16* __restrict__ Qb, u16* __restrict__ Kb, u16* __restrict__ Vt) {
  const int lane = threadIdx.x & 31;
  const int wave = threadIdx.x >> 5;
  const int lh   = lane >> 4;
  const int l    = lane & 15;
  const int m0   = blockIdx.y * 128 + wave * 32;
  const int n0   = blockIdx.x * 64;

  v8f acc[2][4] = {};
  gemm_body(acc, Xb + (m0 + l) * kD, Wb, n0, l, lh);

#pragma unroll
  for (int mt = 0; mt < 2; ++mt) {
#pragma unroll
    for (int j = 0; j < 4; ++j) {
      int en = n0 + j * 16 + l;
#pragma unroll
      for (int e = 0; e < 8; ++e) {
        int gm = m0 + mt * 16 + e + lh * 8;
        int bb = gm >> 11;            // / kS
        int s  = gm & (kS - 1);
        float v = acc[mt][j][e];
        if (en < kD) {                // Q, pre-scaled by 1/sqrt(dk)
          int h = en >> 6, d = en & 63;
          Qb[((bb * kH + h) * kS + s) * kDK + d] = f2bf(v * 0.125f);
        } else if (en < 2 * kD) {     // K
          int e2 = en - kD; int h = e2 >> 6, d = e2 & 63;
          Kb[((bb * kH + h) * kS + s) * kDK + d] = f2bf(v);
        } else {                      // V transposed: [B,H,dk,S]
          int e2 = en - 2 * kD; int h = e2 >> 6, d = e2 & 63;
          Vt[((bb * kH + h) * kDK + d) * kS + s] = f2bf(v);
        }
      }
    }
  }
}

// ---------------------------------------------------------- attention
// One wave = one 16-query tile. K/V key-blocks staged into per-wave
// double-buffered LDS by the Tensor Data Mover; P transposed via LDS.
__global__ __launch_bounds__(128) void attn_kernel(
    const u16* __restrict__ Qb, const u16* __restrict__ Kb,
    const u16* __restrict__ Vt, u16* __restrict__ Cb) {
  // per wave: 2 buffers x (K tile 32x64 + V tile 64x32) = 16KB; 4 waves = 64KB
  __shared__ u16 smem[4 * 2 * 4096];

  const int lane = threadIdx.x & 31;
  const int lh   = lane >> 4;
  const int l    = lane & 15;
  // wave-uniform (SGPR) copy for descriptors / addressing
  const int wave = __builtin_amdgcn_readfirstlane((int)(threadIdx.x >> 5));

  const int wid   = blockIdx.x * 4 + wave;
  const int qt    = wid & 127;
  const int bh    = wid >> 7;
  const int qbase = qt << 4;

  const u16* Qp = Qb + (bh * kS + qbase) * kDK;
  const u16* Kp = Kb + bh * kS * kDK;
  const u16* Vp = Vt + bh * kDK * kS;

  u16* wbase = &smem[wave * 8192];
#define KBUF(bi) (wbase + (bi) * 4096)
#define VBUF(bi) (wbase + (bi) * 4096 + 2048)

  // Q as two A-fragments (dk = 64 -> K-steps 0 and 32)
  Frag16 aq0, aq1;
  {
    const u16* ar = Qp + l * kDK;
    aq0.q[0] = ld4(ar + lh * 16);
    aq0.q[1] = ld4(ar + 8 + lh * 16);
    aq1.q[0] = ld4(ar + 32 + lh * 16);
    aq1.q[1] = ld4(ar + 40 + lh * 16);
  }

  v8f o[4] = {};
  float mi[8], li[8];
#pragma unroll
  for (int e = 0; e < 8; ++e) { mi[e] = -1e30f; li[e] = 0.f; }

#if HAVE_TDM
  // prologue: stage key-block 0 into buffer 0
  tdm_load_tile((u32)(uintptr_t)KBUF(0), Kp, 64, 32, 64, 64, 32);
  tdm_load_tile((u32)(uintptr_t)VBUF(0), Vp, 32, 64, kS, 32, 64);
#endif

  for (int kb = 0; kb <= qbase; kb += 32) {
    const int cur = (kb >> 5) & 1;
    const int nxt = cur ^ 1;

    Frag16 bk[4], bv[4];
#if HAVE_TDM
    if (kb + 32 <= qbase) {   // stage next block while computing this one
      tdm_load_tile((u32)(uintptr_t)KBUF(nxt), Kp + (kb + 32) * kDK, 64, 32, 64, 64, 32);
      tdm_load_tile((u32)(uintptr_t)VBUF(nxt), Vp + (kb + 32), 32, 64, kS, 32, 64);
      WAIT_TENSOR(2);         // 2 newest outstanding = next pair; current pair done
    } else {
      WAIT_TENSOR(0);
    }
    asm volatile("" ::: "memory");
    {
      const u16* kt = KBUF(cur);            // 32 rows x 64 cols
      const u16* vt = VBUF(cur);            // 64 rows x 32 cols
      bk[0].q[0] = ld4(kt + l * 64 + lh * 16);        bk[0].q[1] = ld4(kt + l * 64 + lh * 16 + 8);
      bk[1].q[0] = ld4(kt + l * 64 + 32 + lh * 16);   bk[1].q[1] = ld4(kt + l * 64 + 40 + lh * 16);
      bk[2].q[0] = ld4(kt + (16 + l) * 64 + lh * 16);      bk[2].q[1] = ld4(kt + (16 + l) * 64 + lh * 16 + 8);
      bk[3].q[0] = ld4(kt + (16 + l) * 64 + 32 + lh * 16); bk[3].q[1] = ld4(kt + (16 + l) * 64 + 40 + lh * 16);
#pragma unroll
      for (int j = 0; j < 4; ++j) {
        const u16* vr = vt + (j * 16 + l) * 32 + lh * 16;
        bv[j].q[0] = ld4(vr);
        bv[j].q[1] = ld4(vr + 8);
      }
    }
#else
    {
      const u16* kr0 = Kp + (kb + l) * kDK + lh * 16;
      const u16* kr1 = kr0 + 16 * kDK;
      bk[0].q[0] = ld4(kr0);      bk[0].q[1] = ld4(kr0 + 8);
      bk[1].q[0] = ld4(kr0 + 32); bk[1].q[1] = ld4(kr0 + 40);
      bk[2].q[0] = ld4(kr1);      bk[2].q[1] = ld4(kr1 + 8);
      bk[3].q[0] = ld4(kr1 + 32); bk[3].q[1] = ld4(kr1 + 40);
#pragma unroll
      for (int j = 0; j < 4; ++j) {
        const u16* vr = Vp + (j * 16 + l) * kS + kb + lh * 16;
        bv[j].q[0] = ld4(vr);
        bv[j].q[1] = ld4(vr + 8);
      }
      if (kb + 32 <= qbase) {
        __builtin_prefetch(kr0 + 32 * kDK, 0, 0);
        __builtin_prefetch(kr1 + 32 * kDK, 0, 0);
      }
    }
#endif

    v8f s0 = {}, s1 = {};
    s0 = WMMA_BF16(aq0, bk[0], s0);
    s0 = WMMA_BF16(aq1, bk[1], s0);
    s1 = WMMA_BF16(aq0, bk[2], s1);
    s1 = WMMA_BF16(aq1, bk[3], s1);

    // ---- causal mask + online softmax; P written into the (now free)
    //      current K buffer for the cross-lane A-layout transpose.
    u16* pl = KBUF(cur);
    const int key0 = kb + l;
    const int key1 = kb + 16 + l;
#pragma unroll
    for (int e = 0; e < 8; ++e) {
      int q  = qbase + e + 8 * lh;
      float x0 = (key0 <= q) ? s0[e] : -1e30f;
      float x1 = (key1 <= q) ? s1[e] : -1e30f;
      float mx = fmaxf(x0, x1);
      mx = fmaxf(mx, __shfl_xor(mx, 8, 32));
      mx = fmaxf(mx, __shfl_xor(mx, 4, 32));
      mx = fmaxf(mx, __shfl_xor(mx, 2, 32));
      mx = fmaxf(mx, __shfl_xor(mx, 1, 32));
      float mnew = fmaxf(mi[e], mx);
      float corr = __expf(mi[e] - mnew);
      float p0 = (key0 <= q) ? __expf(x0 - mnew) : 0.f;
      float p1 = (key1 <= q) ? __expf(x1 - mnew) : 0.f;
      float rs = p0 + p1;
      rs += __shfl_xor(rs, 8, 32);
      rs += __shfl_xor(rs, 4, 32);
      rs += __shfl_xor(rs, 2, 32);
      rs += __shfl_xor(rs, 1, 32);
      li[e] = li[e] * corr + rs;
      mi[e] = mnew;
      o[0][e] *= corr; o[1][e] *= corr; o[2][e] *= corr; o[3][e] *= corr;
      int r = e + 8 * lh;
      pl[r * 32 + l]      = f2bf(p0);
      pl[r * 32 + 16 + l] = f2bf(p1);
    }
    asm volatile("s_wait_dscnt 0" ::: "memory");

    Frag16 pa;
    {
      const u16* pr = pl + l * 32;
      pa.q[0] = ld4(pr + lh * 16);
      pa.q[1] = ld4(pr + 8 + lh * 16);
    }
    asm volatile("" ::: "memory");

#pragma unroll
    for (int j = 0; j < 4; ++j) o[j] = WMMA_BF16(pa, bv[j], o[j]);
  }

  // ---- finalize + store context [B,S,H*dk]
  const int b = bh >> 4;
  const int h = bh & 15;
#pragma unroll
  for (int e = 0; e < 8; ++e) {
    float inv = 1.0f / li[e];
    int q = qbase + e + 8 * lh;
    int base = (b * kS + q) * kD + h * kDK;
#pragma unroll
    for (int j = 0; j < 4; ++j)
      Cb[base + j * 16 + l] = f2bf(o[j][e] * inv);
  }
#undef KBUF
#undef VBUF
}

// ------------------------------------------------------------ out GEMM
__global__ __launch_bounds__(128) void gemm_out_kernel(
    const u16* __restrict__ Cb, const u16* __restrict__ Wb,
    float* __restrict__ Out) {
  const int lane = threadIdx.x & 31;
  const int wave = threadIdx.x >> 5;
  const int lh   = lane >> 4;
  const int l    = lane & 15;
  const int m0   = blockIdx.y * 128 + wave * 32;
  const int n0   = blockIdx.x * 64;

  v8f acc[2][4] = {};
  gemm_body(acc, Cb + (m0 + l) * kD, Wb, n0, l, lh);

#pragma unroll
  for (int mt = 0; mt < 2; ++mt) {
#pragma unroll
    for (int j = 0; j < 4; ++j) {
      int en = n0 + j * 16 + l;
#pragma unroll
      for (int e = 0; e < 8; ++e) {
        int gm = m0 + mt * 16 + e + lh * 8;
        Out[gm * kD + en] = acc[mt][j][e];
      }
    }
  }
}

// ------------------------------------------------------------- launcher
extern "C" void kernel_launch(void* const* d_in, const int* in_sizes, int n_in,
                              void* d_out, int out_size, void* d_ws, size_t ws_size,
                              hipStream_t stream) {
  const float* X    = (const float*)d_in[0];
  const float* Wqkv = (const float*)d_in[1];
  const float* Wout = (const float*)d_in[2];
  float* out = (float*)d_out;

  u16* ws  = (u16*)d_ws;
  u16* Xb  = ws;                      // 4,194,304
  u16* Wqb = Xb  + 4194304;           // 3,145,728
  u16* Wob = Wqb + 3145728;           // 1,048,576
  u16* Qb  = Wob + 1048576;           // 4,194,304
  u16* Kb  = Qb  + 4194304;           // 4,194,304
  u16* Vt  = Kb  + 4194304;           // 4,194,304
  u16* Cb  = Vt  + 4194304;           // 4,194,304   (total 48 MB)

  convert_kernel<<<32768, 256, 0, stream>>>(X, Wqkv, Wout, Xb, Wqb, Wob);
  gemm_qkv_kernel<<<dim3(kE / 64, kBS / 128), 128, 0, stream>>>(Xb, Wqb, Qb, Kb, Vt);
  attn_kernel<<<(kB * kH * (kS / 16)) / 4, 128, 0, stream>>>(Qb, Kb, Vt, Cb);
  gemm_out_kernel<<<dim3(kD / 64, kBS / 128), 128, 0, stream>>>(Cb, Wob, out);
}
